// SimpleMambaBlock_7782480740854
// MI455X (gfx1250) — compile-verified, matrix operand order checked by reading, not run on hardware
//
#include <hip/hip_runtime.h>
#include <hip/hip_bf16.h>

#define D_MODEL 512
#define D_STATE 16
#define D_CONVW 4
#define D_INNER 1024
#define DT_RANK 32
#define SEQ     1024
#define ROWS    2048              // B * L
#define D_FFN   2048

typedef __attribute__((ext_vector_type(16))) __bf16 bf16x16;
typedef __attribute__((ext_vector_type(8)))  float  f32x8;

union Frag { bf16x16 v; unsigned int u[8]; };

__device__ __forceinline__ unsigned short f2bf(float f) {
    unsigned int u = __float_as_uint(f);
    u += 0x7FFFu + ((u >> 16) & 1u);            // round-to-nearest-even
    return (unsigned short)(u >> 16);
}
__device__ __forceinline__ float sigmoidf_(float x) { return 1.0f / (1.0f + __expf(-x)); }
__device__ __forceinline__ float siluf_(float x)    { return x * sigmoidf_(x); }
__device__ __forceinline__ float geluf_(float x) {  // tanh approximation (jax.nn.gelu default)
    float x3 = x * x * x;
    return 0.5f * x * (1.0f + tanhf(0.7978845608028654f * (x + 0.044715f * x3)));
}
__device__ __forceinline__ float softplusf_(float x) {
    return (x > 20.0f) ? x : log1pf(__expf(x));
}

// ---- CDNA5 async global->LDS copy (ASYNCcnt-tracked), via inline asm --------
__device__ __forceinline__ void async_copy_b64(unsigned lds_off, const void* gsrc) {
    asm volatile("global_load_async_to_lds_b64 %0, %1, off"
                 :: "v"(lds_off), "v"(gsrc) : "memory");
}
__device__ __forceinline__ void wait_async_le1() {
    asm volatile("s_wait_asynccnt 0x1" ::: "memory");
}
__device__ __forceinline__ void wait_async_le0() {
    asm volatile("s_wait_asynccnt 0x0" ::: "memory");
}

// ---------------------------------------------------------------- f32 -> bf16
__global__ void cvt_f32_bf16(const float* __restrict__ in,
                             unsigned short* __restrict__ out, int n) {
    int i = blockIdx.x * blockDim.x + threadIdx.x;
    if (i < n) out[i] = f2bf(in[i]);
}

// -------------------------------------------------- LayerNorm over 512, bf16 out
__global__ void layernorm512_bf16(const float* __restrict__ x,
                                  const float* __restrict__ g,
                                  const float* __restrict__ b,
                                  unsigned short* __restrict__ out) {
    __shared__ float s0[256], s1[256];
    int row = blockIdx.x, tid = threadIdx.x;
    const float* xr = x + (size_t)row * D_MODEL;
    float v0 = xr[tid], v1 = xr[tid + 256];
    s0[tid] = v0 + v1;
    s1[tid] = v0 * v0 + v1 * v1;
    __syncthreads();
    for (int s = 128; s > 0; s >>= 1) {
        if (tid < s) { s0[tid] += s0[tid + s]; s1[tid] += s1[tid + s]; }
        __syncthreads();
    }
    float mu  = s0[0] * (1.0f / D_MODEL);
    float var = s1[0] * (1.0f / D_MODEL) - mu * mu;
    float rs  = rsqrtf(var + 1e-5f);
    unsigned short* orow = out + (size_t)row * D_MODEL;
    orow[tid]       = f2bf((v0 - mu) * rs * g[tid]       + b[tid]);
    orow[tid + 256] = f2bf((v1 - mu) * rs * g[tid + 256] + b[tid + 256]);
}

// ------------------------------------------------------------------ WMMA GEMM
// C[M,N] = act(A[M,K](bf16) * W[N,K]^T(bf16) + bias) + res
// Block: 256 threads = 8 waves, one 32-row M-slab (blockIdx.y), 8x 64-col wave
// tiles along N (blockIdx.x). A-tile (32x32 bf16) is staged into LDS with
// double-buffered async copies shared by all waves; B fragments load direct
// from global (L2-resident weights). Per wave per K-step: 8x v_wmma.
enum { ACT_NONE = 0, ACT_SOFTPLUS = 1, ACT_GELU = 2 };

template <int ACT, bool BIAS, bool RES, bool SF32, bool SBF16>
__global__ __launch_bounds__(256)
void wmma_gemm(const unsigned short* __restrict__ A, int lda,
               const unsigned short* __restrict__ W, int ldw,
               const float* __restrict__ bias,
               const float* __restrict__ res,
               float* __restrict__ Cf, unsigned short* __restrict__ Cb,
               int ldc, int N, int K) {
    __shared__ unsigned short As[2][32][32];     // 2 x 2KB double buffer
    int wid  = threadIdx.x >> 5;
    int lane = threadIdx.x & 31;
    int tm   = blockIdx.y << 5;
    int tn   = (blockIdx.x * 8 + wid) << 6;
    bool active = (tn < N);                      // wave-uniform
    int hi = lane >> 4;
    int mn = lane & 15;

    // cooperative A staging: thread t copies 8 bytes of row (t>>3)
    int srow = threadIdx.x >> 3;
    int scol = (threadIdx.x & 7) << 2;           // element offset (4 bf16 = 8B)
    const unsigned short* Asrc = A + (size_t)(tm + srow) * lda + scol;
    unsigned lds0 = (unsigned)(unsigned long long)(void*)&As[0][srow][scol];
    unsigned lds1 = (unsigned)(unsigned long long)(void*)&As[1][srow][scol];

    const unsigned short* Wrow[4];
#pragma unroll
    for (int j = 0; j < 4; ++j)
        Wrow[j] = W + (size_t)(tn + 16 * j + mn) * ldw;

    f32x8 acc[2][4];
#pragma unroll
    for (int im = 0; im < 2; ++im)
#pragma unroll
        for (int j = 0; j < 4; ++j) acc[im][j] = {};

    async_copy_b64(lds0, Asrc);                  // stage k0 = 0 into buf 0

    int nsteps = K >> 5;
    for (int s = 0; s < nsteps; ++s) {
        int buf = s & 1;
        bool has_next = (s + 1 < nsteps);
        if (has_next)                            // stage next K-slab into other buf
            async_copy_b64(buf ? lds0 : lds1, Asrc + ((s + 1) << 5));

        // B fragments from global (issue early, overlap with wait)
        Frag b[4];
        if (active) {
#pragma unroll
            for (int j = 0; j < 4; ++j)
#pragma unroll
                for (int i = 0; i < 8; ++i) {
                    int kk = (s << 5) + ((i >> 2) << 4) + (hi << 3) + ((i & 3) << 1);
                    b[j].u[i] = *(const unsigned int*)(Wrow[j] + kk);
                }
        }

        if (has_next) wait_async_le1(); else wait_async_le0();
        __syncthreads();                         // A-tile for step s visible

        Frag a0, a1;
        if (active) {
#pragma unroll
            for (int i = 0; i < 8; ++i) {
                int kk = ((i >> 2) << 4) + (hi << 3) + ((i & 3) << 1);
                a0.u[i] = *(const unsigned int*)&As[buf][mn][kk];
                a1.u[i] = *(const unsigned int*)&As[buf][mn + 16][kk];
            }
#pragma unroll
            for (int j = 0; j < 4; ++j) {
                acc[0][j] = __builtin_amdgcn_wmma_f32_16x16x32_bf16(false, a0.v, false, b[j].v, (short)0, acc[0][j], false, false);
                acc[1][j] = __builtin_amdgcn_wmma_f32_16x16x32_bf16(false, a1.v, false, b[j].v, (short)0, acc[1][j], false, false);
            }
        }
        __syncthreads();                         // done reading buf before reuse
    }

    if (!active) return;
#pragma unroll
    for (int im = 0; im < 2; ++im)
#pragma unroll
        for (int j = 0; j < 4; ++j)
#pragma unroll
            for (int r = 0; r < 8; ++r) {
                int m0 = tm + im * 16 + hi * 8 + r;   // C/D: vgpr r -> row r (+8 upper half)
                int n0 = tn + j * 16 + mn;
                float v = acc[im][j][r];
                if (BIAS) v += bias[n0];
                if (ACT == ACT_SOFTPLUS) v = softplusf_(v);
                else if (ACT == ACT_GELU) v = geluf_(v);
                size_t off = (size_t)m0 * ldc + n0;
                if (RES) v += res[off];
                if (SF32)  Cf[off] = v;
                if (SBF16) Cb[off] = f2bf(v);
            }
}

// ----------------------------------------- depthwise causal conv1d + SiLU gate
__global__ void conv_silu(const float* __restrict__ xz,       // [ROWS, 2*D_INNER]
                          const float* __restrict__ cw,       // [D_INNER, 4]
                          const float* __restrict__ cb,       // [D_INNER]
                          float* __restrict__ xcf,            // [ROWS, D_INNER]
                          unsigned short* __restrict__ xcb) { // [ROWS, D_INNER]
    int row = blockIdx.x;                  // b*SEQ + l
    int b = row >> 10, l = row & 1023;
#pragma unroll
    for (int j = 0; j < 4; ++j) {
        int e = threadIdx.x + (j << 8);
        float acc = cb[e];
#pragma unroll
        for (int k = 0; k < D_CONVW; ++k) {
            int ls = l - (D_CONVW - 1) + k;
            if (ls >= 0)
                acc += xz[((size_t)(b << 10) + ls) * (2 * D_INNER) + e] * cw[e * D_CONVW + k];
        }
        acc = siluf_(acc);
        size_t o = (size_t)row * D_INNER + e;
        xcf[o] = acc;
        xcb[o] = f2bf(acc);
    }
}

// ------------------------------------------------- selective scan (sequential L)
// one thread per (b, e); 16 SSM states live in registers; B/C staged via LDS.
__global__ void ssm_scan(const float* __restrict__ delta,   // [ROWS, D_INNER]
                         const float* __restrict__ xdbl,    // [ROWS, 64]
                         const float* __restrict__ xc,      // [ROWS, D_INNER]
                         const float* __restrict__ xz,      // [ROWS, 2*D_INNER] (z = cols 1024..)
                         const float* __restrict__ A_log,   // [D_INNER, 16]
                         const float* __restrict__ Dv,      // [D_INNER]
                         unsigned short* __restrict__ yb) { // [ROWS, D_INNER]
    __shared__ float sBC[2 * D_STATE];
    int b = blockIdx.x >> 2;
    int e = ((blockIdx.x & 3) << 8) + threadIdx.x;
    float Ae[D_STATE], h[D_STATE];
#pragma unroll
    for (int n = 0; n < D_STATE; ++n) {
        Ae[n] = -__expf(A_log[e * D_STATE + n]);
        h[n] = 0.0f;
    }
    float De = Dv[e];
    for (int l = 0; l < SEQ; ++l) {
        int row = (b << 10) + l;
        if (threadIdx.x < 2 * D_STATE)
            sBC[threadIdx.x] = xdbl[(size_t)row * 64 + DT_RANK + threadIdx.x];
        __syncthreads();
        float dl = delta[(size_t)row * D_INNER + e];
        float xv = xc[(size_t)row * D_INNER + e];
        float y = 0.0f;
#pragma unroll
        for (int n = 0; n < D_STATE; ++n) {
            float dA = __expf(dl * Ae[n]);
            h[n] = dA * h[n] + dl * sBC[n] * xv;
            y += h[n] * sBC[D_STATE + n];
        }
        y += xv * De;
        float zv = xz[(size_t)row * (2 * D_INNER) + D_INNER + e];
        y *= siluf_(zv);
        yb[(size_t)row * D_INNER + e] = f2bf(y);
        __syncthreads();
    }
}

// ---------------------------------------------------------------------- driver
static inline dim3 gemm_grid(int M, int N) {
    return dim3(((N >> 6) + 7) / 8, M >> 5);
}

extern "C" void kernel_launch(void* const* d_in, const int* in_sizes, int n_in,
                              void* d_out, int out_size, void* d_ws, size_t ws_size,
                              hipStream_t stream) {
    (void)in_sizes; (void)n_in; (void)out_size; (void)ws_size;
    const float* x          = (const float*)d_in[0];
    const float* ln1_g      = (const float*)d_in[1];
    const float* ln1_b      = (const float*)d_in[2];
    const float* in_proj_w  = (const float*)d_in[3];
    const float* conv_w     = (const float*)d_in[4];
    const float* conv_b     = (const float*)d_in[5];
    const float* x_proj_w   = (const float*)d_in[6];
    const float* dt_proj_w  = (const float*)d_in[7];
    const float* dt_proj_b  = (const float*)d_in[8];
    const float* A_log      = (const float*)d_in[9];
    const float* Dvec       = (const float*)d_in[10];
    const float* out_proj_w = (const float*)d_in[11];
    const float* ln2_g      = (const float*)d_in[12];
    const float* ln2_b      = (const float*)d_in[13];
    const float* ffn1_w     = (const float*)d_in[14];
    const float* ffn1_b     = (const float*)d_in[15];
    const float* ffn2_w     = (const float*)d_in[16];
    const float* ffn2_b     = (const float*)d_in[17];
    float* out = (float*)d_out;

    char* ws = (char*)d_ws;
    size_t off = 0;
    auto alloc = [&](size_t bytes) -> void* {
        void* p = ws + off;
        off = (off + bytes + 255) & ~(size_t)255;
        return p;
    };

    unsigned short* wb_in  = (unsigned short*)alloc((size_t)(2 * D_INNER) * D_MODEL * 2);
    unsigned short* wb_xp  = (unsigned short*)alloc((size_t)64 * D_INNER * 2);
    unsigned short* wb_dt  = (unsigned short*)alloc((size_t)D_INNER * DT_RANK * 2);
    unsigned short* wb_out = (unsigned short*)alloc((size_t)D_MODEL * D_INNER * 2);
    unsigned short* wb_f1  = (unsigned short*)alloc((size_t)D_FFN * D_MODEL * 2);
    unsigned short* wb_f2  = (unsigned short*)alloc((size_t)D_MODEL * D_FFN * 2);
    unsigned short* xn_b   = (unsigned short*)alloc((size_t)ROWS * D_MODEL * 2);
    float*          xz_f   = (float*)alloc((size_t)ROWS * 2 * D_INNER * 4);
    float*          xc_f   = (float*)alloc((size_t)ROWS * D_INNER * 4);
    unsigned short* xc_b   = (unsigned short*)alloc((size_t)ROWS * D_INNER * 2);
    float*          xdbl_f = (float*)alloc((size_t)ROWS * 64 * 4);
    unsigned short* xdbl_b = (unsigned short*)alloc((size_t)ROWS * 64 * 2);
    float*          dlt_f  = (float*)alloc((size_t)ROWS * D_INNER * 4);
    unsigned short* y_b    = (unsigned short*)alloc((size_t)ROWS * D_INNER * 2);
    float*          xres_f = (float*)alloc((size_t)ROWS * D_MODEL * 4);
    unsigned short* f_b    = (unsigned short*)alloc((size_t)ROWS * D_MODEL * 2);
    unsigned short* g_b    = (unsigned short*)alloc((size_t)ROWS * D_FFN * 2);

    // weight conversions f32 -> bf16
    struct { const float* src; unsigned short* dst; int n; } cv[6] = {
        { in_proj_w,  wb_in,  2 * D_INNER * D_MODEL },
        { x_proj_w,   wb_xp,  64 * D_INNER },
        { dt_proj_w,  wb_dt,  D_INNER * DT_RANK },
        { out_proj_w, wb_out, D_MODEL * D_INNER },
        { ffn1_w,     wb_f1,  D_FFN * D_MODEL },
        { ffn2_w,     wb_f2,  D_MODEL * D_FFN },
    };
    for (int i = 0; i < 6; ++i)
        cvt_f32_bf16<<<(cv[i].n + 255) / 256, 256, 0, stream>>>(cv[i].src, cv[i].dst, cv[i].n);

    // 1) LN1 -> xn (bf16)
    layernorm512_bf16<<<ROWS, 256, 0, stream>>>(x, ln1_g, ln1_b, xn_b);

    // 2) in_proj: xz = xn @ in_proj_w^T   [2048 x 2048], K=512
    wmma_gemm<ACT_NONE, false, false, true, false><<<gemm_grid(ROWS, 2 * D_INNER), 256, 0, stream>>>(
        xn_b, D_MODEL, wb_in, D_MODEL, nullptr, nullptr, xz_f, nullptr,
        2 * D_INNER, 2 * D_INNER, D_MODEL);

    // 3) depthwise causal conv + SiLU -> xc (f32 + bf16)
    conv_silu<<<ROWS, 256, 0, stream>>>(xz_f, conv_w, conv_b, xc_f, xc_b);

    // 4) x_proj: x_dbl = xc @ x_proj_w^T  [2048 x 64], K=1024
    wmma_gemm<ACT_NONE, false, false, true, true><<<gemm_grid(ROWS, 64), 256, 0, stream>>>(
        xc_b, D_INNER, wb_xp, D_INNER, nullptr, nullptr, xdbl_f, xdbl_b,
        64, 64, D_INNER);

    // 5) dt_proj + softplus: delta = softplus(dt @ dt_proj_w^T + b)  [2048 x 1024], K=32
    wmma_gemm<ACT_SOFTPLUS, true, false, true, false><<<gemm_grid(ROWS, D_INNER), 256, 0, stream>>>(
        xdbl_b, 64, wb_dt, DT_RANK, dt_proj_b, nullptr, dlt_f, nullptr,
        D_INNER, D_INNER, DT_RANK);

    // 6) selective scan + skip + SiLU gate -> y (bf16)
    ssm_scan<<<8, 256, 0, stream>>>(dlt_f, xdbl_f, xc_f, xz_f, A_log, Dvec, y_b);

    // 7) out_proj + residual: x_res = x + y @ out_proj_w^T  [2048 x 512], K=1024
    wmma_gemm<ACT_NONE, false, true, true, false><<<gemm_grid(ROWS, D_MODEL), 256, 0, stream>>>(
        y_b, D_INNER, wb_out, D_INNER, nullptr, x, xres_f, nullptr,
        D_MODEL, D_MODEL, D_INNER);

    // 8) LN2 -> f (bf16)
    layernorm512_bf16<<<ROWS, 256, 0, stream>>>(xres_f, ln2_g, ln2_b, f_b);

    // 9) ffn1 + bias + gelu -> g (bf16)  [2048 x 2048], K=512
    wmma_gemm<ACT_GELU, true, false, false, true><<<gemm_grid(ROWS, D_FFN), 256, 0, stream>>>(
        f_b, D_MODEL, wb_f1, D_MODEL, ffn1_b, nullptr, nullptr, g_b,
        D_FFN, D_FFN, D_MODEL);

    // 10) ffn2 + bias + residual -> out (f32)  [2048 x 512], K=2048
    wmma_gemm<ACT_NONE, true, true, true, false><<<gemm_grid(ROWS, D_MODEL), 256, 0, stream>>>(
        g_b, D_FFN, wb_f2, D_FFN, ffn2_b, xres_f, out, nullptr,
        D_MODEL, D_MODEL, D_FFN);
}
